// PretrainedGINForPropertyPrediction_18451179504223
// MI455X (gfx1250) — matrix operand order repaired and checked by reading, not compiled, not run on hardware
//
#include <hip/hip_runtime.h>

typedef __attribute__((ext_vector_type(16))) _Float16 v16h;
typedef __attribute__((ext_vector_type(8)))  _Float16 v8h;
typedef __attribute__((ext_vector_type(8)))  float    v8f;
typedef __attribute__((ext_vector_type(4)))  unsigned int u32x4;
typedef __attribute__((ext_vector_type(8)))  int i32x8;
typedef __attribute__((ext_vector_type(4)))  int i32x4;

#define NN 200000
#define EE 800000
#define DD 128
#define HH 256
#define GG 10000
#define CC 12

#if defined(__gfx1250__) && __has_builtin(__builtin_amdgcn_tensor_load_to_lds)
#define USE_TDM 1
#else
#define USE_TDM 0
#endif

static __device__ __forceinline__ v16h concat8(v8h lo, v8h hi) {
  return __builtin_shufflevector(lo, hi, 0, 1, 2, 3, 4, 5, 6, 7, 8, 9, 10, 11, 12, 13, 14, 15);
}

// ---------------- copy (vectorized float4) ----------------
__global__ void copy4_kernel(const float* __restrict__ src, float* __restrict__ dst, long long n4) {
  long long i = (long long)blockIdx.x * blockDim.x + threadIdx.x;
  if (i < n4) ((float4*)dst)[i] = ((const float4*)src)[i];
}

// ---------------- scatter-add: z[dst] += h[src], f32 global atomics ----------------
__global__ void scatter_add_kernel(const float* __restrict__ h,
                                   const long long* __restrict__ srcI,
                                   const long long* __restrict__ dstI,
                                   float* __restrict__ z, int E, int Hc) {
  const int per = Hc >> 2;  // float4 chunks per edge row
  long long tid = (long long)blockIdx.x * blockDim.x + threadIdx.x;
  if (tid >= (long long)E * per) return;
  int e = (int)(tid / per);
  int q = (int)(tid % per) * 4;
  long long s = srcI[e], d = dstI[e];
  const float4 v = *(const float4*)(h + s * (long long)Hc + q);
  float* p = z + d * (long long)Hc + q;
  atomicAdd(p + 0, v.x);
  atomicAdd(p + 1, v.y);
  atomicAdd(p + 2, v.z);
  atomicAdd(p + 3, v.w);
}

// ---------------- WMMA GEMM: out = act(A[M,K] @ W[K,Nout] + bias) ----------------
// 256 threads = 8 waves. Block tile 256x64; wave tile 32x64 (2 row-frags x 4 col-frags,
// 8 x v_wmma_f32_16x16x32_f16 per K-step).
// A tile (256x32 f32) is DMA'd into LDS by the Tensor Data Mover (TENSORcnt) with
// pad_enable giving a 36-dword row stride (conflict-free ds_load_b128 fragment reads);
// f32->f16 conversion happens in registers. W tile is staged transposed in f16.
#define AROW_STRIDE 36  // 32 data dwords + 4 pad dwords per row (TDM pad: interval=32dw, amount=4dw)

__launch_bounds__(256)
__global__ void gemm_bias_relu_wmma(const float* __restrict__ A, const float* __restrict__ W,
                                    const float* __restrict__ bias, float* __restrict__ out,
                                    int M, int K, int Nout, int relu) {
  __shared__ float    As32[256 * AROW_STRIDE];  // [row][k] f32, padded stride
  __shared__ _Float16 WsT[64][40];              // [n][k] transposed W tile (32 + 8 pad)

  const int tid  = threadIdx.x;
  const int lane = tid & 31;
  const int wave = tid >> 5;
  const int m0 = blockIdx.x * 256;
  const int n0 = blockIdx.y * 64;

  v8f acc[2][4] = {};

  // ISA wave32 operand packing (CDNA5 ISA 7.12.2):
  // A (16x32 f16): lane<16 -> row=lane, K {0..7,16..23}; lane>=16 -> K {8..15,24..31}
  const int am  = lane & 15;
  const int akb = (lane >> 4) * 8;   // 0 or 8
  // B (32x16 f16): lane<16 -> col=lane, K {0..7,16..23}; lane>=16 -> K {8..15,24..31}
  const int bn  = lane & 15;
  const int bkb = (lane >> 4) * 8;   // 0 or 8

#if USE_TDM
  // ---- invariant parts of the Tensor DMA descriptor (D#) ----
  const unsigned int lds_base = (unsigned int)(unsigned long long)(void*)&As32[0];
  const int rem = M - m0;  // remaining rows; TDM zero-fills OOB rows
  i32x8 g1 = {};
  g1[0] = (2 << 16)        // data_size = 4B
        | (1 << 20)        // pad_enable
        | (4 << 22)        // pad_interval: 32 dwords
        | (3 << 25);       // pad_amount: 4 dwords
  g1[1] = (K & 0xffff) << 16;                        // tensor_dim0[15:0]
  g1[2] = ((K >> 16) & 0xffff) | ((rem & 0xffff) << 16);  // dim0 hi | dim1 lo
  g1[3] = ((rem >> 16) & 0xffff) | (32 << 16);       // dim1 hi | tile_dim0 = 32
  g1[4] = 256;                                       // tile_dim1 = 256 rows
  g1[5] = K;                                         // tensor_dim0_stride
  const i32x4 g2 = {}, g3 = {};
#endif

  for (int k0 = 0; k0 < K; k0 += 32) {
#if USE_TDM
    // ---- async DMA of A tile into LDS (wave 0 issues; TENSORcnt tracked) ----
    if (wave == 0) {
      const unsigned long long gaddr =
          (unsigned long long)(const void*)(A + (long long)m0 * K + k0);
      u32x4 g0 = {};
      g0[0] = 1u;                                        // count = 1 valid descriptor
      g0[1] = lds_base;                                  // lds_addr
      g0[2] = (unsigned int)gaddr;                       // global_addr[31:0]
      g0[3] = (unsigned int)(gaddr >> 32) | (2u << 30);  // global_addr[56:32] | type=2
#if __clang_major__ >= 23
      const i32x8 g4 = {};
      __builtin_amdgcn_tensor_load_to_lds(g0, g1, g2, g3, g4, 0);
#else
      __builtin_amdgcn_tensor_load_to_lds(g0, g1, g2, g3, 0);
#endif
    }
#else
    // ---- fallback: manual staging of A tile (f32, padded stride) ----
    {
      const int c0 = (tid & 1) * 16;
#pragma unroll
      for (int p = 0; p < 2; ++p) {
        const int r  = (tid >> 1) + p * 128;
        const int gr = m0 + r;
        float4 v0 = {}, v1 = {}, v2 = {}, v3 = {};
        if (gr < M) {
          const float4* ap = (const float4*)(A + (long long)gr * K + k0 + c0);
          v0 = ap[0]; v1 = ap[1]; v2 = ap[2]; v3 = ap[3];
        }
        float4* sp = (float4*)&As32[r * AROW_STRIDE + c0];
        sp[0] = v0; sp[1] = v1; sp[2] = v2; sp[3] = v3;
      }
    }
#endif
    // ---- stage W tile transposed: WsT[n][k] = W[k0+k][n0+n] ----
    // lane-coalesced global reads (consecutive lanes -> consecutive n)
    {
      const int n  = tid & 63;
      const int kc = (tid >> 6) * 8;  // 0,8,16,24
      const float* wp = W + (long long)(k0 + kc) * Nout + n0 + n;
      v8h v;
#pragma unroll
      for (int i = 0; i < 8; ++i) v[i] = (_Float16)wp[(long long)i * Nout];
      *(v8h*)&WsT[n][kc] = v;
    }
    // prefetch next W K-tile (global_prefetch_b8)
    if (k0 + 32 < K) {
      __builtin_prefetch(W + (long long)(k0 + 32) * Nout + n0 + (tid & 63), 0, 1);
    }
#if USE_TDM
    if (wave == 0) __builtin_amdgcn_s_wait_tensorcnt(0);
#endif
    __syncthreads();

    // ---- A fragments: conflict-free b128 LDS loads + in-register f32->f16 ----
    v16h af[2];
#pragma unroll
    for (int p = 0; p < 2; ++p) {
      const int arow = wave * 32 + p * 16 + am;
      const float* rp = &As32[arow * AROW_STRIDE];
      v16h f;
#pragma unroll
      for (int j = 0; j < 8; ++j) f[j] = (_Float16)rp[akb + j];
#pragma unroll
      for (int j = 0; j < 8; ++j) f[8 + j] = (_Float16)rp[16 + akb + j];
      af[p] = f;
    }

    // ---- B fragments (shared across both row-frags) + 8 WMMAs ----
#pragma unroll
    for (int nt = 0; nt < 4; ++nt) {
      const int brow = nt * 16 + bn;
      v8h lo = *(const v8h*)&WsT[brow][bkb];
      v8h hi = *(const v8h*)&WsT[brow][bkb + 16];
      v16h bf = concat8(lo, hi);
#pragma unroll
      for (int p = 0; p < 2; ++p) {
        acc[p][nt] = __builtin_amdgcn_wmma_f32_16x16x32_f16(
            /*neg_a=*/false, af[p], /*neg_b=*/false, bf,
            /*c_mod=*/(short)0, acc[p][nt], /*reuse_a=*/false, /*reuse_b=*/false);
      }
    }
    __syncthreads();  // protect LDS from next iteration's DMA/stores
  }

  // ---- epilogue: C/D layout -> VGPR r = row M=r (lanes 0-15) / M=r+8 (lanes 16-31) ----
#pragma unroll
  for (int p = 0; p < 2; ++p) {
    const int rowbase = m0 + wave * 32 + p * 16 + ((lane < 16) ? 0 : 8);
#pragma unroll
    for (int nt = 0; nt < 4; ++nt) {
      const int col = n0 + nt * 16 + (lane & 15);
      const float bv = bias[col];
#pragma unroll
      for (int r = 0; r < 8; ++r) {
        const int grow = rowbase + r;
        if (grow < M) {
          float v = acc[p][nt][r] + bv;
          if (relu) v = fmaxf(v, 0.0f);
          out[(long long)grow * Nout + col] = v;
        }
      }
    }
  }
}

// ---------------- BatchNorm statistics ----------------
__global__ void bn_zero_kernel(float* sums, float* sqs) {
  sums[threadIdx.x] = 0.0f;
  sqs[threadIdx.x]  = 0.0f;
}

__global__ void bn_stats_kernel(const float* __restrict__ h, float* __restrict__ sums,
                                float* __restrict__ sqs, int Nrows) {
  const int j = threadIdx.x;  // column 0..255
  float s = 0.0f, q = 0.0f;
  for (long long r = blockIdx.x; r < Nrows; r += gridDim.x) {
    const float v = h[r * HH + j];
    s += v;
    q += v * v;
  }
  atomicAdd(&sums[j], s);
  atomicAdd(&sqs[j], q);
}

__global__ void bn_final_kernel(const float* __restrict__ sums, const float* __restrict__ sqs,
                                const float* __restrict__ gamma, const float* __restrict__ beta,
                                float* __restrict__ scale, float* __restrict__ shift, int Nrows) {
  const int j = threadIdx.x;
  const float invN = 1.0f / (float)Nrows;
  const float mean = sums[j] * invN;
  const float var  = sqs[j] * invN - mean * mean;
  const float sc   = gamma[j] * rsqrtf(var + 1e-5f);
  scale[j] = sc;
  shift[j] = beta[j] - mean * sc;
}

// ---------------- fused BN-apply + per-graph max/mean pool ----------------
// batch is sorted: one block per graph, binary-search its contiguous row range.
__global__ void pool_kernel(const float* __restrict__ h, const long long* __restrict__ batch,
                            const float* __restrict__ scale, const float* __restrict__ shift,
                            float* __restrict__ gfeat, int Nrows) {
  const int g = blockIdx.x;
  int lo, hi;
  {
    int a = 0, b = Nrows;
    while (a < b) { int m = (a + b) >> 1; if (batch[m] < (long long)g) a = m + 1; else b = m; }
    lo = a;
    b = Nrows;
    while (a < b) { int m = (a + b) >> 1; if (batch[m] < (long long)(g + 1)) a = m + 1; else b = m; }
    hi = a;
  }
  const int j = threadIdx.x;  // column
  const float sc = scale[j], sh = shift[j];
  float mx = -3.402823466e38f, sm = 0.0f;
  for (int r = lo; r < hi; ++r) {
    const float v = h[(long long)r * HH + j] * sc + sh;
    mx = fmaxf(mx, v);
    sm += v;
  }
  const int cnt = hi - lo;
  gfeat[(long long)g * (2 * HH) + j]      = (cnt > 0) ? mx : 0.0f;
  gfeat[(long long)g * (2 * HH) + HH + j] = (cnt > 0) ? sm / (float)cnt : 0.0f;
}

// ---------------- head fc2: [G,256] @ [256,12] + b ----------------
__global__ void fc2_kernel(const float* __restrict__ hid, const float* __restrict__ w,
                           const float* __restrict__ b, float* __restrict__ out) {
  const int idx = blockIdx.x * blockDim.x + threadIdx.x;
  if (idx >= GG * CC) return;
  const int g = idx / CC, c = idx % CC;
  const float* hp = hid + (long long)g * HH;
  float s = b[c];
#pragma unroll 8
  for (int j = 0; j < HH; ++j) s += hp[j] * w[j * CC + c];
  out[idx] = s;
}

// =======================================================================
extern "C" void kernel_launch(void* const* d_in, const int* in_sizes, int n_in,
                              void* d_out, int out_size, void* d_ws, size_t ws_size,
                              hipStream_t stream) {
  const float*      x    = (const float*)d_in[0];
  const long long*  ei   = (const long long*)d_in[1];  // [2,E] row-major
  const long long*  bat  = (const long long*)d_in[2];
  const float* w1[3] = { (const float*)d_in[3],  (const float*)d_in[7],  (const float*)d_in[11] };
  const float* b1[3] = { (const float*)d_in[4],  (const float*)d_in[8],  (const float*)d_in[12] };
  const float* w2[3] = { (const float*)d_in[5],  (const float*)d_in[9],  (const float*)d_in[13] };
  const float* b2[3] = { (const float*)d_in[6],  (const float*)d_in[10], (const float*)d_in[14] };
  const float* bn_gamma = (const float*)d_in[15];
  const float* bn_beta  = (const float*)d_in[16];
  const float* fc1_w = (const float*)d_in[17];
  const float* fc1_b = (const float*)d_in[18];
  const float* fc2_w = (const float*)d_in[19];
  const float* fc2_b = (const float*)d_in[20];
  float* out = (float*)d_out;

  const long long* srcI = ei;        // edge_index[0]
  const long long* dstI = ei + EE;   // edge_index[1]

  // workspace carving
  const long long NH = (long long)NN * HH;
  float* z     = (float*)d_ws;       // [N,128] or [N,256]
  float* y     = z + NH;             // [N,256]
  float* h     = y + NH;             // [N,256]
  float* sums  = h + NH;             // [256]
  float* sqs   = sums + HH;          // [256]
  float* scale = sqs + HH;           // [256]
  float* shift = scale + HH;         // [256]
  float* gfeat = shift + HH;         // [G,512]
  float* hid   = gfeat + (long long)GG * 2 * HH;  // [G,256]

  const dim3 gemmBlk(256);
  const int mTiles = (NN + 255) / 256;

  // ---------------- layer 0 (K = 128) ----------------
  {
    const long long n4 = (long long)NN * DD / 4;
    copy4_kernel<<<(int)((n4 + 255) / 256), 256, 0, stream>>>(x, z, n4);
    const long long st = (long long)EE * (DD / 4);
    scatter_add_kernel<<<(int)((st + 255) / 256), 256, 0, stream>>>(x, srcI, dstI, z, EE, DD);
    gemm_bias_relu_wmma<<<dim3(mTiles, HH / 64), gemmBlk, 0, stream>>>(z, w1[0], b1[0], y, NN, DD, HH, 1);
    gemm_bias_relu_wmma<<<dim3(mTiles, HH / 64), gemmBlk, 0, stream>>>(y, w2[0], b2[0], h, NN, HH, HH, 1);
  }
  // ---------------- layers 1, 2 (K = 256) ----------------
  for (int l = 1; l < 3; ++l) {
    const long long n4 = NH / 4;
    copy4_kernel<<<(int)((n4 + 255) / 256), 256, 0, stream>>>(h, z, n4);
    const long long st = (long long)EE * (HH / 4);
    scatter_add_kernel<<<(int)((st + 255) / 256), 256, 0, stream>>>(h, srcI, dstI, z, EE, HH);
    gemm_bias_relu_wmma<<<dim3(mTiles, HH / 64), gemmBlk, 0, stream>>>(z, w1[l], b1[l], y, NN, HH, HH, 1);
    gemm_bias_relu_wmma<<<dim3(mTiles, HH / 64), gemmBlk, 0, stream>>>(y, w2[l], b2[l], h, NN, HH, HH, 1);
  }

  // ---------------- BatchNorm (training-mode batch stats) ----------------
  bn_zero_kernel<<<1, HH, 0, stream>>>(sums, sqs);
  bn_stats_kernel<<<512, HH, 0, stream>>>(h, sums, sqs, NN);
  bn_final_kernel<<<1, HH, 0, stream>>>(sums, sqs, bn_gamma, bn_beta, scale, shift, NN);

  // ---------------- fused BN-apply + max/mean pool ----------------
  pool_kernel<<<GG, HH, 0, stream>>>(h, bat, scale, shift, gfeat, NN);

  // ---------------- head ----------------
  const int gTiles = (GG + 255) / 256;
  gemm_bias_relu_wmma<<<dim3(gTiles, HH / 64), gemmBlk, 0, stream>>>(gfeat, fc1_w, fc1_b, hid, GG, 2 * HH, HH, 1);
  fc2_kernel<<<(GG * CC + 255) / 256, 256, 0, stream>>>(hid, fc2_w, fc2_b, out);
}
